// MaskGeneratorNet_16312285790732
// MI455X (gfx1250) — matrix operand order repaired and checked by reading, compiled, not run on hardware
//
#include <hip/hip_runtime.h>
#include <hip/hip_bf16.h>

#define D       4096
#define FOURD   16384
#define T_STEPS 200
#define XDIM    64
#define EMBN    128
#define H1N     1024
#define KTOP    2048   // D * (1 - 0.5)

typedef __attribute__((ext_vector_type(16))) __bf16 v16bf;
typedef __attribute__((ext_vector_type(8)))  __bf16 v8bf;
typedef __attribute__((ext_vector_type(8)))  float  v8f;

// ---------- one-time f32 -> bf16 conversion of Whh (256MB -> 128MB, fits 192MB L2) ----------
__global__ void cvt_bf16_kernel(const float* __restrict__ src, __bf16* __restrict__ dst, int n) {
    int i = blockIdx.x * blockDim.x + threadIdx.x;
    int stride = gridDim.x * blockDim.x;
    for (; i < n; i += stride) dst[i] = (__bf16)src[i];
}

// ---------- precompute xWih[t][r] = x[t] . Wih[r]  (batched, 0.2 GMAC) ----------
__global__ void xwih_kernel(const float* __restrict__ x, const float* __restrict__ Wih,
                            float* __restrict__ xwih) {
    int idx = blockIdx.x * blockDim.x + threadIdx.x;
    if (idx >= T_STEPS * FOURD) return;
    int t = idx / FOURD, r = idx - t * FOURD;
    const float4* xr = (const float4*)(x + (size_t)t * XDIM);
    const float4* wr = (const float4*)(Wih + (size_t)r * XDIM);
    float s = 0.f;
#pragma unroll
    for (int k = 0; k < XDIM / 4; ++k) {
        float4 a = xr[k], b = wr[k];
        s += a.x * b.x + a.y * b.y + a.z * b.z + a.w * b.w;
    }
    xwih[idx] = s;
}

// ---------- zero initial z (= Whh @ h0 = 0) and c ----------
__global__ void lstm_init(float* __restrict__ z0, float* __restrict__ c0) {
    int j = blockIdx.x * blockDim.x + threadIdx.x;
    if (j < FOURD) z0[j] = 0.f;
    if (j < D)     c0[j] = 0.f;
}

// ---------- fused LSTM step ----------
// Phase 1 (every block, redundantly): h_{t+1} = gates(z_in + xwih[t] + b, c_in),
//   written straight into LDS as bf16 (the WMMA A-fragment source); a second
//   D-sized LDS region is zero-filled so non-row-0 lanes fetch zeros by
//   ADDRESS instead of needing VALU cndmask zeroing (kills the WMMA->VALU
//   hazard NOPs in the inner loop). Block 0 persists c_{t+1} and h_{t+1}.
// Phase 2: z_out = Whh_bf16 @ h_{t+1} via WMMA, N padded to 16.
//   B tile (32x16): lane l<16 -> column N=l, K=0..15 contiguous from a Whh row;
//   lanes 16-31 -> K=16..31.  A tile (16x32): row 0 = h chunk (lanes 0 & 16
//   hold data per the 16-bit A layout), other rows read the zero region.
__global__ void __launch_bounds__(256)
lstm_step_fused(const __bf16* __restrict__ whh, const float* __restrict__ z_in,
                const float* __restrict__ xw_t, const float* __restrict__ b,
                const float* __restrict__ c_in, float* __restrict__ c_out,
                float* __restrict__ z_out, float* __restrict__ h_out) {
    __shared__ __align__(16) __bf16 sh[2 * D];   // [0,D): h_bf16, [D,2D): zeros

    // ---- phase 1: gate update (uniform trip count, no divergence) ----
    const bool writer = (blockIdx.x == 0);
    for (int j = threadIdx.x; j < D; j += 256) {
        float iv = z_in[j]         + xw_t[j]         + b[j];
        float fv = z_in[D + j]     + xw_t[D + j]     + b[D + j];
        float gv = z_in[2 * D + j] + xw_t[2 * D + j] + b[2 * D + j];
        float ov = z_in[3 * D + j] + xw_t[3 * D + j] + b[3 * D + j];
        iv = 1.f / (1.f + __expf(-iv));
        fv = 1.f / (1.f + __expf(-fv));
        ov = 1.f / (1.f + __expf(-ov));
        gv = tanhf(gv);
        float cv = fv * c_in[j] + iv * gv;
        float hv = ov * tanhf(cv);
        sh[j] = (__bf16)hv;
        sh[D + j] = (__bf16)0.f;
        if (writer) { c_out[j] = cv; h_out[j] = hv; }
    }
    __syncthreads();

    // ---- phase 2: WMMA GEMV ----
    const int lane = threadIdx.x & 31;
    const int wave = threadIdx.x >> 5;
    const int row0 = blockIdx.x * 128 + wave * 16;
    const int mrow = lane & 15;
    const int half = lane >> 4;            // 0: K 0..15, 1: K 16..31 of each chunk
    const bool keep = (mrow == 0);         // only A-matrix row 0 (lanes 0,16) holds h
    const __bf16* brow = whh + (size_t)(row0 + mrow) * D + half * 16;
    // non-keep lanes read from the zero region: selection by address, not VALU
    const __bf16* abase = sh + (keep ? 0 : D) + half * 16;

    v8f c0 = {}, c1 = {};
    for (int k0 = 0; k0 < D; k0 += 64) {
        v16bf b0 = *(const v16bf*)(brow + k0);
        union { v16bf v; v8bf h[2]; } ua0;
        ua0.h[0] = *(const v8bf*)(abase + k0);
        ua0.h[1] = *(const v8bf*)(abase + k0 + 8);
        c0 = __builtin_amdgcn_wmma_f32_16x16x32_bf16(false, ua0.v, false, b0,
                                                     (short)0, c0, false, false);
        v16bf b1 = *(const v16bf*)(brow + k0 + 32);
        union { v16bf v; v8bf h[2]; } ua1;
        ua1.h[0] = *(const v8bf*)(abase + k0 + 32);
        ua1.h[1] = *(const v8bf*)(abase + k0 + 40);
        c1 = __builtin_amdgcn_wmma_f32_16x16x32_bf16(false, ua1.v, false, b1,
                                                     (short)0, c1, false, false);
    }
    // D row M=0 lives in accumulator element 0, lanes 0..15 (N = lane)
    float acc = c0[0] + c1[0];
    if (lane < 16) z_out[row0 + lane] = acc;
}

// ---------- em_base MLP ----------
__global__ void em1_kernel(const float* __restrict__ W1, const float* __restrict__ b1,
                           const float* __restrict__ emb_in, float* __restrict__ h1out) {
    int r = blockIdx.x * blockDim.x + threadIdx.x;
    if (r >= H1N) return;
    const float4* wr = (const float4*)(W1 + (size_t)r * EMBN);
    const float4* xv = (const float4*)emb_in;
    float s = 0.f;
#pragma unroll 8
    for (int k = 0; k < EMBN / 4; ++k) {
        float4 a = wr[k], b = xv[k];
        s += a.x * b.x + a.y * b.y + a.z * b.z + a.w * b.w;
    }
    s += b1[r];
    h1out[r] = fmaxf(s, 0.f);
}

__global__ void em2_kernel(const float* __restrict__ W2, const float* __restrict__ b2,
                           const float* __restrict__ h1v, const float* __restrict__ hlstm,
                           float* __restrict__ embedding, float* __restrict__ vecA) {
    int r = blockIdx.x * blockDim.x + threadIdx.x;
    if (r >= D) return;
    const float4* wr = (const float4*)(W2 + (size_t)r * H1N);
    const float4* xv = (const float4*)h1v;
    float s = 0.f;
    for (int k = 0; k < H1N / 4; ++k) {
        float4 a = wr[k], b = xv[k];
        s += a.x * b.x + a.y * b.y + a.z * b.z + a.w * b.w;
    }
    float e = (s + b2[r]) * hlstm[r];
    embedding[r] = e;
    vecA[r] = fmaxf(e, 0.f);   // relu(embedding) feeds fc0
}

// ---------- 4096x4096 GEMV, one wave per row ----------
// mode 0: y = W@v + b          mode 1: y = relu((W@v + b) * emb)
__global__ void __launch_bounds__(256)
gemv4096(const float* __restrict__ W, const float* __restrict__ bias,
         const float* __restrict__ v, const float* __restrict__ emb,
         float* __restrict__ y, int mode) {
    int lane = threadIdx.x & 31;
    int row = blockIdx.x * 8 + (threadIdx.x >> 5);
    const float4* wr = (const float4*)(W + (size_t)row * D);
    const float4* vv = (const float4*)v;
    float s = 0.f;
    for (int k = lane; k < D / 4; k += 32) {
        float4 a = wr[k], b = vv[k];
        s += a.x * b.x + a.y * b.y + a.z * b.z + a.w * b.w;
    }
#pragma unroll
    for (int o = 16; o > 0; o >>= 1) s += __shfl_xor(s, o, 32);
    if (lane == 0) {
        s += bias[row];
        if (mode == 1) { s *= emb[row]; s = fmaxf(s, 0.f); }
        y[row] = s;
    }
}

// ---------- bound: raw = (t - min) / (max - min), write probs row + next input ----------
__global__ void bound_kernel(const float* __restrict__ t, float* __restrict__ raw,
                             float* __restrict__ probs_row) {
    __shared__ float smn[256], smx[256];
    float mn = 3.4e38f, mx = -3.4e38f;
    for (int j = threadIdx.x; j < D; j += 256) {
        float v = t[j]; mn = fminf(mn, v); mx = fmaxf(mx, v);
    }
    smn[threadIdx.x] = mn; smx[threadIdx.x] = mx;
    __syncthreads();
    for (int s2 = 128; s2 > 0; s2 >>= 1) {
        if (threadIdx.x < s2) {
            smn[threadIdx.x] = fminf(smn[threadIdx.x], smn[threadIdx.x + s2]);
            smx[threadIdx.x] = fmaxf(smx[threadIdx.x], smx[threadIdx.x + s2]);
        }
        __syncthreads();
    }
    float mnv = smn[0];
    float inv = 1.f / (smx[0] - mnv);
    for (int j = threadIdx.x; j < D; j += 256) {
        float r = (t[j] - mnv) * inv;
        raw[j] = r; probs_row[j] = r;
    }
}

// ---------- top-K binarize via threshold bisection (values are in [0,1]) ----------
__global__ void binarize_kernel(const float* __restrict__ probs, float* __restrict__ binary) {
    int m = blockIdx.x;
    const float* v = probs + (size_t)m * D;
    float* o = binary + (size_t)m * D;
    __shared__ int cnt;
    float lo = 0.f, hi = 1.f;
    for (int it = 0; it < 24; ++it) {
        float mid = 0.5f * (lo + hi);
        if (threadIdx.x == 0) cnt = 0;
        __syncthreads();
        int c = 0;
        for (int j = threadIdx.x; j < D; j += blockDim.x) c += (v[j] > mid) ? 1 : 0;
        atomicAdd(&cnt, c);
        __syncthreads();
        if (cnt > KTOP) lo = mid; else hi = mid;
        __syncthreads();
    }
    float thr = hi;   // count(v > thr) <= K
    for (int j = threadIdx.x; j < D; j += blockDim.x)
        o[j] = (v[j] > thr && v[j] > 0.f) ? 1.f : 0.f;
}

extern "C" void kernel_launch(void* const* d_in, const int* in_sizes, int n_in,
                              void* d_out, int out_size, void* d_ws, size_t ws_size,
                              hipStream_t stream) {
    const float* x      = (const float*)d_in[0];
    const float* emb_in = (const float*)d_in[1];
    const float* Wih    = (const float*)d_in[2];
    const float* Whh    = (const float*)d_in[3];
    const float* lstm_b = (const float*)d_in[4];
    const float* em_W1  = (const float*)d_in[5];
    const float* em_b1  = (const float*)d_in[6];
    const float* em_W2  = (const float*)d_in[7];
    const float* em_b2  = (const float*)d_in[8];
    const float* fc0_W  = (const float*)d_in[9];
    const float* fc0_b  = (const float*)d_in[10];
    const float* cond_W = (const float*)d_in[11];
    const float* cond_b = (const float*)d_in[12];
    const float* fc_W   = (const float*)d_in[13];
    const float* fc_b   = (const float*)d_in[14];

    float* probs  = (float*)d_out;         // [5, D]
    float* binary = probs + 5 * D;         // [5, D]

    // workspace layout (~148 MB)
    char* w = (char*)d_ws;
    __bf16* whh_bf  = (__bf16*)w;  w += (size_t)FOURD * D * sizeof(__bf16);
    float*  xwih    = (float*)w;   w += (size_t)T_STEPS * FOURD * sizeof(float);
    float*  zbuf0   = (float*)w;   w += FOURD * sizeof(float);
    float*  zbuf1   = (float*)w;   w += FOURD * sizeof(float);
    float*  cbuf0   = (float*)w;   w += D * sizeof(float);
    float*  cbuf1   = (float*)w;   w += D * sizeof(float);
    float*  h       = (float*)w;   w += D * sizeof(float);
    float*  h1v     = (float*)w;   w += H1N * sizeof(float);
    float*  embedv  = (float*)w;   w += D * sizeof(float);
    float*  vecA    = (float*)w;   w += D * sizeof(float);
    float*  vecB    = (float*)w;   w += D * sizeof(float);
    float*  tmp     = (float*)w;   w += D * sizeof(float);
    float*  zb[2]   = { zbuf0, zbuf1 };
    float*  cb[2]   = { cbuf0, cbuf1 };
    (void)in_sizes; (void)n_in; (void)out_size; (void)ws_size;

    // 1) Whh -> bf16 (so 200 re-reads are L2-resident), batched x@Wih^T, state init
    cvt_bf16_kernel<<<8192, 256, 0, stream>>>(Whh, whh_bf, FOURD * D);
    xwih_kernel<<<(T_STEPS * FOURD + 255) / 256, 256, 0, stream>>>(x, Wih, xwih);
    lstm_init<<<(FOURD + 255) / 256, 256, 0, stream>>>(zb[0], cb[0]);

    // 2) sequential LSTM: one fused (gate + WMMA GEMV) launch per step,
    //    double-buffered z and c to avoid intra-launch races
    for (int t = 0; t < T_STEPS; ++t) {
        lstm_step_fused<<<FOURD / 128, 256, 0, stream>>>(
            whh_bf, zb[t & 1], xwih + (size_t)t * FOURD, lstm_b,
            cb[t & 1], cb[(t + 1) & 1], zb[(t + 1) & 1], h);
    }

    // 3) em_base MLP; embedding = (mlp) * h_final; vecA = relu(embedding)
    em1_kernel<<<(H1N + 255) / 256, 256, 0, stream>>>(em_W1, em_b1, emb_in, h1v);
    em2_kernel<<<(D + 255) / 256, 256, 0, stream>>>(em_W2, em_b2, h1v, h, embedv, vecA);

    // 4) mask pipeline: fc0 + 4 loop layers, each GEMV -> bound
    gemv4096<<<D / 8, 256, 0, stream>>>(fc0_W, fc0_b, vecA, embedv, tmp, 0);
    bound_kernel<<<1, 256, 0, stream>>>(tmp, vecA, probs + 0 * D);
    for (int i = 0; i < 4; ++i) {
        gemv4096<<<D / 8, 256, 0, stream>>>(cond_W + (size_t)i * D * D, cond_b + i * D,
                                            vecA, embedv, vecB, 1);
        gemv4096<<<D / 8, 256, 0, stream>>>(fc_W + (size_t)i * D * D, fc_b + i * D,
                                            vecB, embedv, tmp, 0);
        bound_kernel<<<1, 256, 0, stream>>>(tmp, vecA, probs + (size_t)(i + 1) * D);
    }

    // 5) top-K binarization of the 5 prob rows
    binarize_kernel<<<5, 256, 0, stream>>>(probs, binary);
}
